// Net_gaussian_correction_with_sampling_76630806495763
// MI455X (gfx1250) — matrix operand order repaired and checked by reading, not compiled, not run on hardware
//
#include <hip/hip_runtime.h>
#include <hip/hip_bf16.h>
#include <math.h>

typedef __attribute__((ext_vector_type(2)))  float    v2f;
typedef __attribute__((ext_vector_type(8)))  float    v8f;
typedef __attribute__((ext_vector_type(16))) _Float16 v16h;
typedef __attribute__((ext_vector_type(4)))  unsigned int u32x4;
typedef __attribute__((ext_vector_type(8)))  int      i32x8;
typedef __attribute__((ext_vector_type(4)))  int      i32x4;

#define NNODES 32768
#define FEAT   128

#if defined(__has_builtin)
#if __has_builtin(__builtin_amdgcn_wmma_f32_16x16x4_f32)
#define USE_WMMA_F32 1
#endif
#if __has_builtin(__builtin_amdgcn_tensor_load_to_lds) && __has_builtin(__builtin_amdgcn_s_wait_tensorcnt)
#define HAVE_TDM 1
#endif
#endif

__device__ __forceinline__ float sigmoidf(float x){ return 1.0f/(1.0f+__expf(-x)); }

#if defined(HAVE_TDM)
// 1-D TDM copy: nelem f32 from contiguous global to contiguous LDS.
// D# packing per CDNA5 ISA (group0/group1 bitfields).
__device__ __forceinline__ void tdm_load_1d(const float* g, float* lds, unsigned nelem){
  unsigned long long ga = (unsigned long long)(size_t)g;
  unsigned lo = (unsigned)(size_t)lds;           // flat LDS addr low 32 = byte offset
  u32x4 g0;
  g0[0] = 1u;                                    // count=1, is_restore=0, gather off
  g0[1] = lo;                                    // lds_addr      [63:32]
  g0[2] = (unsigned)(ga & 0xffffffffull);        // global_addr   [95:64]
  g0[3] = (unsigned)((ga >> 32) & 0x1ffffffull)  // global_addr   [120:96]
        | (2u << 30);                            // type=2 (image)[127:126]
  i32x8 g1;
  g1[0] = (int)(2u << 16);                       // wg_mask=0, data_size=2 (4B)
  g1[1] = (int)((nelem & 0xffffu) << 16);        // tensor_dim0[15:0]  @bits63:48
  g1[2] = (int)(((nelem >> 16) & 0xffffu)        // tensor_dim0[31:16] @bits79:64
        | (1u << 16));                           // tensor_dim1=1      @bits95:80
  g1[3] = (int)((nelem & 0xffffu) << 16);        // tile_dim0          @bits127:112
  g1[4] = 0;                                     // tile_dim1=0, tile_dim2=0
  g1[5] = (int)nelem;                            // tensor_dim0_stride low32
  g1[6] = 0;
  g1[7] = 0;
  i32x4 g2 = {0,0,0,0}, g3 = {0,0,0,0};
#if __clang_major__ >= 23
  i32x8 g4 = {0,0,0,0,0,0,0,0};
  __builtin_amdgcn_tensor_load_to_lds(g0, g1, g2, g3, g4, 0);
#else
  __builtin_amdgcn_tensor_load_to_lds(g0, g1, g2, g3, 0);
#endif
}
#endif

// ---------------- lin0: h = sigmoid(x @ lin0_w.T), zero-pad 64..127 ----------------
__global__ void lin0_kernel(const float* __restrict__ x, const float* __restrict__ w,
                            float* __restrict__ h){
  int tid = blockIdx.x*blockDim.x + threadIdx.x;   // NNODES*128 threads
  int n = tid >> 7, c = tid & 127;
  float acc = 0.f;
  if (c < 64){
    #pragma unroll
    for (int k = 0; k < 32; ++k) acc += x[(size_t)n*32 + k] * w[(size_t)c*32 + k];
    acc = sigmoidf(acc);
  }
  h[(size_t)n*FEAT + c] = (c < 64) ? acc : 0.f;
}

// ---------------- WMMA GEMM: C[N x NC] = A[N x 128] @ B (+bias) ----------------
// Block = 256 threads (8 waves) computes a 16-row x 128-col tile of C.
// A panel (16x128, 8KB) and B panel (128x128, 64KB) staged in LDS via TDM.
// TRANSB: B element (k,c) = B[c*128+k]; the needed 128-row panel of B is contiguous.
template<bool TRANSB, bool HASBIAS>
__global__ void wmma_gemm_kernel(const float* __restrict__ A, const float* __restrict__ B,
                                 const float* __restrict__ bias, float* __restrict__ C,
                                 int ldc)
{
  __shared__ float sA[16*128];     // 8KB
  __shared__ float sB[128*128];    // 64KB
  const int lane = threadIdx.x & 31;
  const int wave = threadIdx.x >> 5;
  const int row0 = blockIdx.x * 16;
  const int col0 = blockIdx.y * 128 + wave * 16;
  const int half = lane >> 4;      // 0: lanes 0-15, 1: lanes 16-31
  const int l    = lane & 15;
  const int K = 128;

  const float* Apanel = A + (size_t)row0 * K;                       // 16x128 contiguous
  const float* Bpanel = TRANSB ? (B + (size_t)(blockIdx.y*128)*K)   // 128 rows of W
                               : B;                                 // full 128x128 W
#if defined(HAVE_TDM)
  if (threadIdx.x < 32) {          // wave 0 issues both DMAs (wave-level, EXEC ignored)
    tdm_load_1d(Apanel, sA, 16*128);
    tdm_load_1d(Bpanel, sB, 128*128);
    __builtin_amdgcn_s_wait_tensorcnt(0);
  }
  __syncthreads();
#else
  for (int i = threadIdx.x; i < (16*128)/4; i += 256)
    reinterpret_cast<float4*>(sA)[i] = reinterpret_cast<const float4*>(Apanel)[i];
  for (int i = threadIdx.x; i < (128*128)/4; i += 256)
    reinterpret_cast<float4*>(sB)[i] = reinterpret_cast<const float4*>(Bpanel)[i];
  __syncthreads();
#endif

  v8f acc = {};
#if defined(USE_WMMA_F32)
  for (int k = 0; k < K; k += 4) {
    const int kb = k + 2*half;
    v2f a;
    { const float2 t = *reinterpret_cast<const float2*>(&sA[l*K + kb]);
      a.x = t.x; a.y = t.y; }
    v2f b;
    if (TRANSB) {
      const float2 t = *reinterpret_cast<const float2*>(&sB[(wave*16 + l)*K + kb]);
      b.x = t.x; b.y = t.y;
    } else {
      b.x = sB[kb*K     + wave*16 + l];
      b.y = sB[(kb+1)*K + wave*16 + l];
    }
    acc = __builtin_amdgcn_wmma_f32_16x16x4_f32(false, a, false, b, (short)0, acc,
                                                false, false);
  }
#else
  for (int k = 0; k < K; k += 32) {
    v16h a, b;
    #pragma unroll
    for (int e = 0; e < 16; ++e) {
      const int ka = half*16 + e;
      a[e] = (_Float16)sA[l*K + k + ka];
      const int kb = half*16 + e;
      const float bl = TRANSB ? sB[(wave*16 + l)*K + k + kb]
                              : sB[(k+kb)*K + wave*16 + l];
      b[e] = (_Float16)bl;
    }
    acc = __builtin_amdgcn_wmma_f32_16x16x32_f16(false, a, false, b, (short)0, acc,
                                                 false, false);
  }
#endif
  float bv = 0.f;
  if (HASBIAS) bv = bias[col0 + l];
  #pragma unroll
  for (int r = 0; r < 8; ++r) {
    const int row = row0 + r + 8*half;            // C layout: lanes>=16 carry M=8..15
    C[(size_t)row*ldc + col0 + l] = acc[r] + bv;
  }
}

// ---------------- edge scatter: agg[dst] += m[src] ----------------
__global__ void zero_kernel(float* __restrict__ p, int n){
  int i = blockIdx.x*blockDim.x + threadIdx.x;
  if (i < n) p[i] = 0.f;
}

__global__ void scatter_kernel(const float* __restrict__ m, const int* __restrict__ src,
                               const int* __restrict__ dst, float* __restrict__ agg, int E){
  int gid = blockIdx.x*blockDim.x + threadIdx.x;
  int e = gid >> 5;
  if (e >= E) return;
  int f = (gid & 31) * 4;
  int s = src[e] & (NNODES - 1);
  int d = dst[e] & (NNODES - 1);
  const float4 v = *reinterpret_cast<const float4*>(m + (size_t)s*FEAT + f);
  float* o = agg + (size_t)d*FEAT + f;
  atomicAdd(o + 0, v.x); atomicAdd(o + 1, v.y);
  atomicAdd(o + 2, v.z); atomicAdd(o + 3, v.w);
}

// ---------------- GRU gate math (elementwise) ----------------
__global__ void gru_gates_kernel(const float* __restrict__ gi, const float* __restrict__ gh,
                                 float* __restrict__ h){
  int idx = blockIdx.x*blockDim.x + threadIdx.x;  // NNODES*128
  int n = idx >> 7, j = idx & 127;
  size_t b = (size_t)n*384;
  float r  = sigmoidf(gi[b + j]       + gh[b + j]);
  float z  = sigmoidf(gi[b + 128 + j] + gh[b + 128 + j]);
  float nn = tanhf   (gi[b + 256 + j] + r*gh[b + 256 + j]);
  size_t hi = (size_t)n*FEAT + j;
  h[hi] = (1.f - z)*nn + z*h[hi];
}

// ---------------- head: relu + mu/sigma dot products (wave32 reductions) ----------------
__global__ void head_kernel(const float* __restrict__ h,
                            const float* __restrict__ w1, const float* __restrict__ b1,
                            const float* __restrict__ w2, const float* __restrict__ b2,
                            float* __restrict__ mu, float* __restrict__ sg){
  int lane = threadIdx.x & 31;
  int wave = threadIdx.x >> 5;
  int node = blockIdx.x*8 + wave;
  float a1 = 0.f, a2 = 0.f;
  for (int k = lane; k < FEAT; k += 32){
    float hv = fmaxf(h[(size_t)node*FEAT + k], 0.f);
    a1 += hv * w1[k];
    a2 += hv * w2[k];
  }
  #pragma unroll
  for (int off = 16; off > 0; off >>= 1){
    a1 += __shfl_xor(a1, off, 32);
    a2 += __shfl_xor(a2, off, 32);
  }
  if (lane == 0){
    mu[node] = a1 + b1[0];
    float q = a2 + b2[0];
    sg[node] = (q > 20.f) ? q : log1pf(__expf(q));   // softplus
  }
}

// ---------------- sampler: closed-form Cholesky of diag(s) - s s^T/denom ----------------
__device__ __forceinline__ float blk_incl_scan(float v, float* sh){
  int t = threadIdx.x;
  sh[t] = v; __syncthreads();
  #pragma unroll
  for (int off = 1; off < 256; off <<= 1){
    float add = (t >= off) ? sh[t - off] : 0.f;
    __syncthreads();
    sh[t] += add;
    __syncthreads();
  }
  float r = sh[t]; __syncthreads();
  return r;
}

__device__ __forceinline__ float blk_sum(float v, float* sh){
  int t = threadIdx.x;
  sh[t] = v; __syncthreads();
  #pragma unroll
  for (int off = 128; off > 0; off >>= 1){
    if (t < off) sh[t] += sh[t + off];
    __syncthreads();
  }
  float r = sh[0]; __syncthreads();
  return r;
}

__device__ __forceinline__ float gauss_hash(unsigned g, unsigned i){
  unsigned s0 = (g * 2654435761u) ^ (i * 40503u) ^ 123u;
  unsigned a  = s0 * 747796405u + 2891336453u;
  unsigned b  = ((a >> ((a >> 28) + 4u)) ^ a) * 277803737u;
  unsigned u1 = (b >> 22) ^ b;
  unsigned a2 = (s0 ^ 0x9E3779B9u) * 747796405u + 2891336453u;
  unsigned b2 = ((a2 >> ((a2 >> 28) + 4u)) ^ a2) * 277803737u;
  unsigned u2 = (b2 >> 22) ^ b2;
  float f1 = ((float)u1 + 1.0f) * 2.3283064e-10f;   // (0,1]
  float f2 = (float)u2 * 2.3283064e-10f;
  return sqrtf(-2.0f * __logf(f1)) * __cosf(6.2831853f * f2);
}

__global__ void sample_kernel(const float* __restrict__ mu, const float* __restrict__ sigma,
                              float* __restrict__ out){
  __shared__ float sAm[256];
  __shared__ float sBm[256];
  __shared__ float s_tot, s_siglast, s_mulast;
  const int g = blockIdx.x, t = threadIdx.x;

  float sig = sigma[(size_t)g*256 + t];
  float m   = mu[(size_t)g*256 + t];
  float s   = (t < 255) ? sig : 0.f;

  float inc = blk_incl_scan(s, sAm);                // inclusive prefix of s
  if (t == 255){ s_tot = inc; s_siglast = sig; s_mulast = m; }
  __syncthreads();
  const float denom = s_tot + s_siglast;

  const float ex   = inc - s;                       // exclusive prefix
  const float r_j  = denom - ex;                    // sigma_last + sum_{k>=j} s_k
  const float r_j1 = r_j - s;                       // r_{j+1}
  const float alpha_next = -1.f / r_j1;             // alpha_{j+1}
  const float dprime = s * r_j1 / r_j;              // d'_j  (Cholesky diag^2)
  const float ljj = sqrtf(fmaxf(dprime, 0.f));

  // rm = cov @ rhs without materializing cov:
  const float c = (0.0f /*K*/ - s_mulast) / s_siglast;
  const float srhs = (t < 255) ? (c*s + m) : 0.f;   // s_i * rhs_i
  const float sum_srhs = blk_sum(srhs, sBm);
  const float rm = srhs - s * (sum_srhs / denom);

  // X = rm + L z with L_ij = s_i * alpha_{j+1} * sqrt(d'_j) (i>j), L_jj = sqrt(d'_j)
  const float z = (t < 255) ? gauss_hash((unsigned)g, (unsigned)t) : 0.f;
  const float w = (t < 255) ? alpha_next * ljj * z : 0.f;
  const float incw = blk_incl_scan(w, sAm);
  const float Sw = incw - w;                        // exclusive prefix of w
  const float X = (t < 255) ? (rm + ljj*z + s*Sw) : 0.f;

  const float sumX = blk_sum(X, sBm);
  if (t < 255) out[(size_t)g*256 + t] = X;
  else         out[(size_t)g*256 + 255] = 0.0f - sumX;   // last = K - sum(X)
}

// ---------------- launch ----------------
extern "C" void kernel_launch(void* const* d_in, const int* in_sizes, int n_in,
                              void* d_out, int out_size, void* d_ws, size_t ws_size,
                              hipStream_t stream)
{
  (void)n_in; (void)out_size; (void)ws_size;
  const float* x      = (const float*)d_in[0];
  const float* lin0_w = (const float*)d_in[1];
  const float* ggc_w  = (const float*)d_in[2];
  const float* w_ih   = (const float*)d_in[3];
  const float* w_hh   = (const float*)d_in[4];
  const float* b_ih   = (const float*)d_in[5];
  const float* b_hh   = (const float*)d_in[6];
  const float* lin1_w = (const float*)d_in[7];
  const float* lin1_b = (const float*)d_in[8];
  const float* lin2_w = (const float*)d_in[9];
  const float* lin2_b = (const float*)d_in[10];
  const int*   eidx   = (const int*)d_in[11];
  const int E = in_sizes[11] / 2;

  float* ws  = (float*)d_ws;
  float* h   = ws;
  float* m   = h   + (size_t)NNODES*FEAT;
  float* agg = m   + (size_t)NNODES*FEAT;
  float* gi  = agg + (size_t)NNODES*FEAT;
  float* gh  = gi  + (size_t)NNODES*384;
  float* muv = gh  + (size_t)NNODES*384;
  float* sgv = muv + NNODES;
  float* out = (float*)d_out;

  dim3 blk(256);
  lin0_kernel<<<NNODES*FEAT/256, blk, 0, stream>>>(x, lin0_w, h);

  dim3 g128(NNODES/16, 1);   // 16-row tiles x 128 cols per block
  dim3 g384(NNODES/16, 3);   // 384 gate outputs
  for (int i = 0; i < 4; ++i){
    wmma_gemm_kernel<false,false><<<g128, blk, 0, stream>>>(h, ggc_w + (size_t)i*FEAT*FEAT,
                                                            nullptr, m, 128);
    zero_kernel<<<NNODES*FEAT/256, blk, 0, stream>>>(agg, NNODES*FEAT);
    scatter_kernel<<<(E*32)/256, blk, 0, stream>>>(m, eidx, eidx + E, agg, E);
    wmma_gemm_kernel<true,true><<<g384, blk, 0, stream>>>(agg, w_ih, b_ih, gi, 384);
    wmma_gemm_kernel<true,true><<<g384, blk, 0, stream>>>(h,   w_hh, b_hh, gh, 384);
    gru_gates_kernel<<<NNODES*FEAT/256, blk, 0, stream>>>(gi, gh, h);
  }
  head_kernel<<<NNODES/8, blk, 0, stream>>>(h, lin1_w, lin1_b, lin2_w, lin2_b, muv, sgv);
  sample_kernel<<<NNODES/256, blk, 0, stream>>>(muv, sgv, out);
}
